// NN_KarplusModel_89876485636174
// MI455X (gfx1250) — matrix operand order repaired and checked by reading, not compiled, not run on hardware
//
#include <hip/hip_runtime.h>
#include <hip/hip_bf16.h>

// ---------------------------------------------------------------------------
// DDSP Karplus-Strong model, CDNA5 (gfx1250, wave32) implementation.
//   - dense layers via v_wmma_f32_16x16x32_bf16 (bf16 in, f32 accum)
//   - weights stored TRANSPOSED (W^T[N][K]) so a B fragment is two b128
//     loads per lane instead of 16 strided u16 loads
//   - GRU: gi = X@Wih hoisted to a parallel WMMA GEMM; sequential part is a
//     single 32-wave workgroup, one 16x16 WMMA tile row per step (M = batch),
//     with global_prefetch on the re-streamed Whh^T strips
//   - Karplus-Strong comb parallelized in chunks of 128 (< delay 159) with
//     history in an LDS ring buffer
// ---------------------------------------------------------------------------

#define Hh   512
#define Bn   16
#define Tn   250
#define NSn  64000
#define LFn  64
#define EFn  64
#define EOn  16
#define DINTn 159
#define FRACF 0.6f
#define NEXC 352            // excitation support (burst is zero after n>=335)

typedef __bf16 bf16;
typedef __attribute__((ext_vector_type(16))) __bf16 v16bf;
typedef __attribute__((ext_vector_type(8)))  float  v8f;

// ---- WMMA fragment loaders (contiguous k = 16*half + elem in A and B), vectorized ----------------
// A: 16x32 bf16 from row-major [M][ld]. lane&15 = M row; half = lane>>4.
//   elems 0..7  = K (half*16 .. half*16+7)    -> one 16B load
//   elems 8..15 = K (half*16+8 .. half*16+15) -> one 16B load at +16B
__device__ __forceinline__ v16bf ld_afrag(const bf16* base, int ld, int k0, int lane) {
  int m = lane & 15, half = lane >> 4;
  const bf16* p = base + (size_t)m * ld + k0 + half * 16;
  union { uint4 u[2]; v16bf v; } r;
  r.u[0] = *reinterpret_cast<const uint4*>(p);
  r.u[1] = *reinterpret_cast<const uint4*>(p + 8);
  return r.v;
}

// B: 32x16 from TRANSPOSED weights Wt[N][ldt] (ldt = K of original matrix).
// lane&15 = N col; lanes 0-15 hold K=k0..k0+15, lanes 16-31 K=k0+16..k0+31.
// Column-contiguous storage -> 32 contiguous bytes per lane.
__device__ __forceinline__ v16bf ld_bfrag_t(const bf16* Wt, int ldt, int k0, int c0, int lane) {
  int n = lane & 15, half = lane >> 4;
  const uint4* q = reinterpret_cast<const uint4*>(Wt + (size_t)(c0 + n) * ldt + k0 + half * 16);
  union { uint4 u[2]; v16bf v; } r;
  r.u[0] = q[0];
  r.u[1] = q[1];
  return r.v;
}

__device__ __forceinline__ float wred(float v) {
#pragma unroll
  for (int o = 16; o; o >>= 1) v += __shfl_xor(v, o, 32);
  return v;
}

// ---- weight conversions ----------------------------------------------------
__global__ void k_cvt_bf16(const float* __restrict__ s, bf16* __restrict__ d, int n) {
  int i = blockIdx.x * blockDim.x + threadIdx.x;
  if (i < n) d[i] = (bf16)s[i];
}

// f32 [K][N] row-major  ->  bf16 [N][K] (transposed)
__global__ void k_cvt_t(const float* __restrict__ s, bf16* __restrict__ d, int K, int N) {
  int i = blockIdx.x * blockDim.x + threadIdx.x;
  if (i >= K * N) return;
  int k = i / N, n = i - k * N;
  d[(size_t)n * K + k] = (bf16)s[i];
}

// ---- in_mlp layer 1 (din==1): outer product + LN + LeakyReLU --------------
// rows 0..3999 = pitch, 4000..7999 = loudness (shared weights).
__global__ __launch_bounds__(256)
void k_in_first(const float* __restrict__ pitch, const float* __restrict__ loud,
                const float* __restrict__ W1, const float* __restrict__ b1,
                const float* __restrict__ g1, const float* __restrict__ be1,
                bf16* __restrict__ Y) {
  int wid = threadIdx.x >> 5, lane = threadIdx.x & 31;
  int r = blockIdx.x * 8 + wid;                     // 0..7999
  float x = (r < Bn * Tn) ? pitch[r] : loud[r - Bn * Tn];
  float v[16], s = 0.f, s2 = 0.f;
#pragma unroll
  for (int j = 0; j < 16; ++j) {
    int c = lane + 32 * j;
    float y = x * W1[c] + b1[c];
    v[j] = y; s += y; s2 += y * y;
  }
  s = wred(s); s2 = wred(s2);
  float mu = s * (1.f / Hh), var = s2 * (1.f / Hh) - mu * mu;
  float inv = rsqrtf(var + 1e-5f);
#pragma unroll
  for (int j = 0; j < 16; ++j) {
    int c = lane + 32 * j;
    float y = (v[j] - mu) * inv * g1[c] + be1[c];
    y = y > 0.f ? y : 0.01f * y;
    Y[(size_t)r * Hh + c] = (bf16)y;
  }
}

// ---- generic 512-col MLP layer: Y = LeakyReLU(LN(X@W + b [+ extras])) -----
// 512 threads = 16 waves; block computes 16 rows x 512 cols; wave w owns col
// tiles {32w, 32w+16}. Wt is transposed [512][512]. If ex0!=null, rank-2
// update with exW (2x512 bf16 = original W rows 512/513, out_mlp layer 1).
// Output routed via (rowmod, offmul) so the last in_mlp layer writes directly
// into the GRU input Xcat[4000][1024].
__global__ __launch_bounds__(512)
void k_mlp_layer(const bf16* __restrict__ X, int ldx,
                 const bf16* __restrict__ Wt,
                 const float* __restrict__ bias, const float* __restrict__ gamma,
                 const float* __restrict__ beta,
                 const float* __restrict__ ex0, const float* __restrict__ ex1,
                 const bf16* __restrict__ exW,
                 bf16* __restrict__ Y, int ldy, int rowmod, int offmul, int N) {
  __shared__ __align__(16) float smem[16 * Hh];     // 32 KB; low 16 KB = A tile
  bf16* As = reinterpret_cast<bf16*>(smem);
  const int tid = threadIdx.x, lane = tid & 31, wv = tid >> 5;
  const int row0 = blockIdx.x * 16;
  {                                                 // stage A tile 16x512 bf16
    int rr = tid >> 5, c0 = (tid & 31) * 16;
    uint4 z0 = {0, 0, 0, 0}, z1 = {0, 0, 0, 0};
    if (row0 + rr < N) {
      const uint4* src = reinterpret_cast<const uint4*>(X + (size_t)(row0 + rr) * ldx + c0);
      z0 = src[0]; z1 = src[1];
    }
    uint4* dst = reinterpret_cast<uint4*>(As + rr * Hh + c0);
    dst[0] = z0; dst[1] = z1;
  }
  __syncthreads();
  int c0a = 32 * wv, c0b = 32 * wv + 16;
  v8f acc0 = {}, acc1 = {};
#pragma unroll 2
  for (int kk = 0; kk < 16; ++kk) {
    int k0 = kk * 32;
    v16bf a  = ld_afrag(As, Hh, k0, lane);
    v16bf b0 = ld_bfrag_t(Wt, Hh, k0, c0a, lane);
    v16bf b1 = ld_bfrag_t(Wt, Hh, k0, c0b, lane);
    acc0 = __builtin_amdgcn_wmma_f32_16x16x32_bf16(false, a, false, b0, (short)0, acc0, false, false);
    acc1 = __builtin_amdgcn_wmma_f32_16x16x32_bf16(false, a, false, b1, (short)0, acc1, false, false);
  }
  __syncthreads();                                  // done with A tile
  int n = lane & 15, half = lane >> 4;
#pragma unroll
  for (int i = 0; i < 8; ++i) {
    int m = i + 8 * half, rr = row0 + m;
    int ca = c0a + n, cb = c0b + n;
    float va = acc0[i] + bias[ca];
    float vb = acc1[i] + bias[cb];
    if (ex0 && rr < N) {                            // K=514 rank-2 update
      float e0 = ex0[rr], e1 = ex1[rr];
      va += e0 * (float)exW[ca] + e1 * (float)exW[Hh + ca];
      vb += e0 * (float)exW[cb] + e1 * (float)exW[Hh + cb];
    }
    smem[m * Hh + ca] = va;
    smem[m * Hh + cb] = vb;
  }
  __syncthreads();
  {                                                 // LN + LeakyReLU: wave wv = row wv
    int m = wv, rr = row0 + m;
    float s = 0.f, s2 = 0.f, v[16];
#pragma unroll
    for (int j = 0; j < 16; ++j) {
      float x = smem[m * Hh + lane + 32 * j];
      v[j] = x; s += x; s2 += x * x;
    }
    s = wred(s); s2 = wred(s2);
    float mu = s * (1.f / Hh), var = s2 * (1.f / Hh) - mu * mu;
    float inv = rsqrtf(var + 1e-5f);
    if (rr < N) {
      int yr = rr % rowmod, yo = (rr / rowmod) * offmul;
#pragma unroll
      for (int j = 0; j < 16; ++j) {
        int c = lane + 32 * j;
        float y = (v[j] - mu) * inv * gamma[c] + beta[c];
        y = y > 0.f ? y : 0.01f * y;
        Y[(size_t)yr * ldy + yo + c] = (bf16)y;
      }
    }
  }
}

// ---- GI = Xcat[4000][1024] @ Wih[1024][1536] + bih, f32 out ---------------
// Wih stored transposed: [1536][1024].
__global__ __launch_bounds__(1024)
void k_gemm_gi(const bf16* __restrict__ X, const bf16* __restrict__ WihT,
               const float* __restrict__ bih, float* __restrict__ GI, int N) {
  __shared__ __align__(16) bf16 As[16 * 1024];      // 32 KB
  const int tid = threadIdx.x, lane = tid & 31, wv = tid >> 5;
  const int row0 = blockIdx.x * 16;
  {
    int rr = tid >> 6, c0 = (tid & 63) * 16;
    uint4 z0 = {0, 0, 0, 0}, z1 = {0, 0, 0, 0};
    if (row0 + rr < N) {
      const uint4* src = reinterpret_cast<const uint4*>(X + (size_t)(row0 + rr) * 1024 + c0);
      z0 = src[0]; z1 = src[1];
    }
    uint4* dst = reinterpret_cast<uint4*>(As + rr * 1024 + c0);
    dst[0] = z0; dst[1] = z1;
  }
  __syncthreads();
  int c0_0 = 16 * wv, c0_1 = 16 * (wv + 32), c0_2 = 16 * (wv + 64);
  v8f a0 = {}, a1 = {}, a2 = {};
#pragma unroll 2
  for (int kk = 0; kk < 32; ++kk) {
    int k0 = kk * 32;
    v16bf a  = ld_afrag(As, 1024, k0, lane);
    v16bf b0 = ld_bfrag_t(WihT, 1024, k0, c0_0, lane);
    v16bf b1 = ld_bfrag_t(WihT, 1024, k0, c0_1, lane);
    v16bf b2 = ld_bfrag_t(WihT, 1024, k0, c0_2, lane);
    a0 = __builtin_amdgcn_wmma_f32_16x16x32_bf16(false, a, false, b0, (short)0, a0, false, false);
    a1 = __builtin_amdgcn_wmma_f32_16x16x32_bf16(false, a, false, b1, (short)0, a1, false, false);
    a2 = __builtin_amdgcn_wmma_f32_16x16x32_bf16(false, a, false, b2, (short)0, a2, false, false);
  }
  int n = lane & 15, half = lane >> 4;
#pragma unroll
  for (int i = 0; i < 8; ++i) {
    int m = i + 8 * half, rr = row0 + m;
    if (rr < N) {
      GI[(size_t)rr * 1536 + c0_0 + n] = a0[i] + bih[c0_0 + n];
      GI[(size_t)rr * 1536 + c0_1 + n] = a1[i] + bih[c0_1 + n];
      GI[(size_t)rr * 1536 + c0_2 + n] = a2[i] + bih[c0_2 + n];
    }
  }
}

// ---- sequential GRU: one workgroup, 32 waves, 250 steps -------------------
// M=16 (batch) is one WMMA tile row. Wave wv owns columns j in [16wv,16wv+16);
// its three accumulators hold gh at (j, j+512, j+1024) = (r,z,n) gates for the
// SAME j, so gating is purely per-lane. h kept as bf16 tile in LDS.
// WhhT is transposed: [1536][512].
__global__ __launch_bounds__(1024)
void k_gru(const float* __restrict__ GI, const bf16* __restrict__ WhhT,
           const float* __restrict__ bhh, bf16* __restrict__ hs) {
  __shared__ __align__(16) bf16 hbf[Bn * Hh];       // 16 KB
  const int tid = threadIdx.x, lane = tid & 31, wv = tid >> 5;
  for (int i = tid; i < Bn * Hh; i += 1024) hbf[i] = (bf16)0.f;
  __syncthreads();
  const int c0_0 = 16 * wv, c0_1 = c0_0 + 512, c0_2 = c0_0 + 1024;
  const int n = lane & 15, half = lane >> 4;
  for (int t = 0; t < Tn; ++t) {
    v8f a0 = {}, a1 = {}, a2 = {};
#pragma unroll 2
    for (int kk = 0; kk < 16; ++kk) {
      int k0 = kk * 32;
      if (kk + 1 < 16) {                            // prefetch next k-step's strips
        int kn = k0 + 32 + half * 16;               // (serial-path latency hiding)
        __builtin_prefetch(WhhT + (size_t)(c0_0 + n) * Hh + kn, 0, 3);
        __builtin_prefetch(WhhT + (size_t)(c0_1 + n) * Hh + kn, 0, 3);
        __builtin_prefetch(WhhT + (size_t)(c0_2 + n) * Hh + kn, 0, 3);
      }
      v16bf a  = ld_afrag(hbf, Hh, k0, lane);
      v16bf b0 = ld_bfrag_t(WhhT, Hh, k0, c0_0, lane);
      v16bf b1 = ld_bfrag_t(WhhT, Hh, k0, c0_1, lane);
      v16bf b2 = ld_bfrag_t(WhhT, Hh, k0, c0_2, lane);
      a0 = __builtin_amdgcn_wmma_f32_16x16x32_bf16(false, a, false, b0, (short)0, a0, false, false);
      a1 = __builtin_amdgcn_wmma_f32_16x16x32_bf16(false, a, false, b1, (short)0, a1, false, false);
      a2 = __builtin_amdgcn_wmma_f32_16x16x32_bf16(false, a, false, b2, (short)0, a2, false, false);
    }
    __syncthreads();                                // all waves done reading h
    int j = c0_0 + n;
#pragma unroll
    for (int i = 0; i < 8; ++i) {
      int m = i + 8 * half;                         // batch index
      const float* gi = GI + ((size_t)m * Tn + t) * 1536;
      float hr = a0[i] + bhh[j];
      float hz = a1[i] + bhh[512 + j];
      float hn = a2[i] + bhh[1024 + j];
      float r  = 1.f / (1.f + __expf(-(gi[j] + hr)));
      float z  = 1.f / (1.f + __expf(-(gi[512 + j] + hz)));
      float nn = tanhf(gi[1024 + j] + r * hn);
      float hp = (float)hbf[m * Hh + j];
      float hw = (1.f - z) * nn + z * hp;
      hbf[m * Hh + j] = (bf16)hw;
      hs[((size_t)m * Tn + t) * Hh + j] = (bf16)hw;
    }
    __syncthreads();                                // h updated for next step
  }
}

// ---- mean pool over T: hbar[b][j] = mean_t out_mlp(h)[b,t,j] --------------
__global__ void k_pool(const bf16* __restrict__ X, float* __restrict__ hbar) {
  int i = blockIdx.x * blockDim.x + threadIdx.x;
  if (i >= Bn * Hh) return;
  int b = i >> 9, j = i & 511;
  float s = 0.f;
  for (int t = 0; t < Tn; ++t) s += (float)X[((size_t)b * Tn + t) * Hh + j];
  hbar[i] = s * (1.f / Tn);
}

// ---- heads: loop taps (softmax*sigmoid(gain)) + excitation FIR coeffs -----
__global__ __launch_bounds__(256)
void k_heads(const float* __restrict__ hbar,
             const float* __restrict__ loopW, const float* __restrict__ loopB,
             const float* __restrict__ excW,  const float* __restrict__ excB,
             const float* __restrict__ gainW, const float* __restrict__ gainB,
             float* __restrict__ bfr, float* __restrict__ aex) {
  int id = blockIdx.x * blockDim.x + threadIdx.x;
  if (id >= Bn * LFn) return;
  int b = id >> 6, f = id & 63;
  const float* h = hbar + b * Hh;
  float g = gainB[f];
  for (int i = 0; i < Hh; ++i) g += h[i] * gainW[i * LFn + f];
  g = 1.f / (1.f + __expf(-g));
  float lc[3];
#pragma unroll
  for (int k = 0; k < 3; ++k) {
    float s = loopB[f * 3 + k];
    for (int i = 0; i < Hh; ++i) s += h[i] * loopW[i * (LFn * 3) + f * 3 + k];
    lc[k] = s;
  }
  float mx = fmaxf(lc[0], fmaxf(lc[1], lc[2]));
  float e0 = __expf(lc[0] - mx), e1 = __expf(lc[1] - mx), e2 = __expf(lc[2] - mx);
  float den = 1.f / (e0 + e1 + e2);
  float* pb = bfr + (b * LFn + f) * 3;
  pb[0] = e0 * den * g; pb[1] = e1 * den * g; pb[2] = e2 * den * g;
  for (int k = 0; k < EOn; ++k) {
    float s = excB[f * EOn + k];
    for (int i = 0; i < Hh; ++i) s += h[i] * excW[i * (EFn * EOn) + f * EOn + k];
    aex[(b * LFn + f) * EOn + k] = s;
  }
}

// ---- excitation: exc[n,b] = sum_k a_s[b,n,k]*burst[n-k]; burst==0 for n>=320
__global__ void k_excite(const float* __restrict__ aex, const float* __restrict__ burst,
                         float* __restrict__ exc) {
  int id = blockIdx.x * blockDim.x + threadIdx.x;
  if (id >= NEXC * Bn) return;
  int npos = id >> 4, b = id & 15;
  float pos = npos * ((float)(EFn - 1) / (float)(NSn - 1));
  int i0 = (int)pos; if (i0 > EFn - 2) i0 = EFn - 2;
  float w = pos - (float)i0;
  const float* A0 = aex + (size_t)(b * EFn + i0) * EOn;
  const float* A1 = A0 + EOn;
  float s = 0.f;
#pragma unroll
  for (int k = 0; k < EOn; ++k) {
    int m = npos - k;
    if (m >= 0) s += (A0[k] * (1.f - w) + A1[k] * w) * burst[m];
  }
  exc[id] = s;
}

// ---- Karplus-Strong comb: y[n] = e[n] + sum_k b[n,k]*(0.4*y[n-159-k] +
// 0.6*y[n-160-k]). Delay >= 159 > 128 -> chunks of 128 samples are internally
// independent; history lives in a 512-entry LDS ring. One block per batch.
__global__ __launch_bounds__(128)
void k_ks(const float* __restrict__ bfr, const float* __restrict__ exc,
          float* __restrict__ out) {
  __shared__ float ring[512];
  const int b = blockIdx.x, tid = threadIdx.x;
  for (int i = tid; i < 512; i += 128) ring[i] = 0.f;
  __syncthreads();
  const float* bb = bfr + b * LFn * 3;
  for (int c = 0; c < NSn / 128; ++c) {
    int nidx = c * 128 + tid;
    float pos = nidx * ((float)(LFn - 1) / (float)(NSn - 1));
    int i0 = (int)pos; if (i0 > LFn - 2) i0 = LFn - 2;
    float w = pos - (float)i0;
    float y = (nidx < NEXC) ? exc[nidx * Bn + b] : 0.f;
#pragma unroll
    for (int k = 0; k < 3; ++k) {
      float bk = bb[i0 * 3 + k] * (1.f - w) + bb[(i0 + 1) * 3 + k] * w;
      int iA = nidx - DINTn - k, iB = iA - 1;
      float vA = (iA >= 0) ? ring[iA & 511] : 0.f;
      float vB = (iB >= 0) ? ring[iB & 511] : 0.f;
      y += bk * ((1.f - FRACF) * vA + FRACF * vB);
    }
    ring[nidx & 511] = y;
    out[(size_t)b * NSn + nidx] = y;
    __syncthreads();
  }
}

// ---------------------------------------------------------------------------
// Input leaf order (setup_inputs dict insertion order, depth-first):
//  0 pitch[4000] 1 loudness[4000]
//  2..13  in_mlp: (W,b,gamma,beta) x3   (W1 is [1,512])
// 14..25  out_mlp: (W,b,gamma,beta) x3  (W1 is [514,512])
// 26 Wih[1024,1536] 27 Whh[512,1536] 28 bih[1536] 29 bhh[1536]
// 30 loop_W[512,192] 31 loop_b[192] 32 exc_W[512,1024] 33 exc_b[1024]
// 34 gain_W[512,64] 35 gain_b[64] 36 burst[64000]
// ---------------------------------------------------------------------------
extern "C" void kernel_launch(void* const* d_in, const int* in_sizes, int n_in,
                              void* d_out, int out_size, void* d_ws, size_t ws_size,
                              hipStream_t stream) {
  const float* pitch = (const float*)d_in[0];
  const float* loud  = (const float*)d_in[1];

  char* ws = (char*)d_ws;
  size_t off = 0;
  auto take = [&](size_t bytes) -> char* {
    off = (off + 255) & ~(size_t)255;
    char* p = ws + off; off += bytes; return p;
  };

  // bf16 weight copies (all GEMM weights transposed: [N][K])
  bf16* cW2i = (bf16*)take((size_t)Hh * Hh * 2);
  bf16* cW3i = (bf16*)take((size_t)Hh * Hh * 2);
  bf16* cW1o = (bf16*)take((size_t)Hh * Hh * 2);    // rows 0..511 of [514,512]
  bf16* cWex = (bf16*)take((size_t)2 * Hh * 2);     // rows 512,513 (pitch/loud)
  bf16* cW2o = (bf16*)take((size_t)Hh * Hh * 2);
  bf16* cW3o = (bf16*)take((size_t)Hh * Hh * 2);
  bf16* cWih = (bf16*)take((size_t)1024 * 1536 * 2);
  bf16* cWhh = (bf16*)take((size_t)512 * 1536 * 2);
  // activations
  bf16*  actA   = (bf16*)take((size_t)8000 * Hh * 2);
  bf16*  actB   = (bf16*)take((size_t)8000 * Hh * 2);
  bf16*  Xcat   = (bf16*)take((size_t)4000 * 1024 * 2);
  float* GI     = (float*)take((size_t)4000 * 1536 * 4);
  bf16*  hsbuf  = (bf16*)take((size_t)4000 * Hh * 2);
  bf16*  oA     = (bf16*)take((size_t)4000 * Hh * 2);
  bf16*  oB     = (bf16*)take((size_t)4000 * Hh * 2);
  bf16*  oC     = (bf16*)take((size_t)4000 * Hh * 2);
  float* hbar   = (float*)take((size_t)Bn * Hh * 4);
  float* bfr    = (float*)take((size_t)Bn * LFn * 3 * 4);
  float* aex    = (float*)take((size_t)Bn * LFn * EOn * 4);
  float* excBuf = (float*)take((size_t)NEXC * Bn * 4);
  (void)ws_size; (void)n_in; (void)in_sizes; (void)out_size;

  auto cvtT = [&](int idx, bf16* dst, int K, int N) {
    k_cvt_t<<<(K * N + 255) / 256, 256, 0, stream>>>((const float*)d_in[idx], dst, K, N);
  };
  cvtT(6,  cW2i, Hh, Hh);
  cvtT(10, cW3i, Hh, Hh);
  cvtT(14, cW1o, Hh, Hh);                           // reads only rows 0..511
  k_cvt_bf16<<<(2 * Hh + 255) / 256, 256, 0, stream>>>(
      (const float*)d_in[14] + (size_t)Hh * Hh, cWex, 2 * Hh);
  cvtT(18, cW2o, Hh, Hh);
  cvtT(22, cW3o, Hh, Hh);
  cvtT(26, cWih, 1024, 1536);
  cvtT(27, cWhh, 512, 1536);

  // in_mlp (shared weights, 8000 rows = pitch|loudness)
  k_in_first<<<1000, 256, 0, stream>>>(pitch, loud,
      (const float*)d_in[2], (const float*)d_in[3],
      (const float*)d_in[4], (const float*)d_in[5], actA);
  k_mlp_layer<<<500, 512, 0, stream>>>(actA, Hh, cW2i,
      (const float*)d_in[7], (const float*)d_in[8], (const float*)d_in[9],
      nullptr, nullptr, nullptr, actB, Hh, 8000, 0, 8000);
  // layer 3 writes straight into Xcat: rows<4000 -> cols 0..511 (pm),
  // rows>=4000 -> cols 512..1023 (lm)
  k_mlp_layer<<<500, 512, 0, stream>>>(actB, Hh, cW3i,
      (const float*)d_in[11], (const float*)d_in[12], (const float*)d_in[13],
      nullptr, nullptr, nullptr, Xcat, 1024, 4000, 512, 8000);

  // GRU: parallel gi, then sequential scan
  k_gemm_gi<<<250, 1024, 0, stream>>>(Xcat, cWih, (const float*)d_in[28], GI, 4000);
  k_gru<<<1, 1024, 0, stream>>>(GI, cWhh, (const float*)d_in[29], hsbuf);

  // out_mlp (layer1: K=514 via rank-2 extras pitch/loudness)
  k_mlp_layer<<<250, 512, 0, stream>>>(hsbuf, Hh, cW1o,
      (const float*)d_in[15], (const float*)d_in[16], (const float*)d_in[17],
      pitch, loud, cWex, oA, Hh, 4000, 0, 4000);
  k_mlp_layer<<<250, 512, 0, stream>>>(oA, Hh, cW2o,
      (const float*)d_in[19], (const float*)d_in[20], (const float*)d_in[21],
      nullptr, nullptr, nullptr, oB, Hh, 4000, 0, 4000);
  k_mlp_layer<<<250, 512, 0, stream>>>(oB, Hh, cW3o,
      (const float*)d_in[23], (const float*)d_in[24], (const float*)d_in[25],
      nullptr, nullptr, nullptr, oC, Hh, 4000, 0, 4000);

  k_pool<<<(Bn * Hh + 255) / 256, 256, 0, stream>>>(oC, hbar);
  k_heads<<<(Bn * LFn + 255) / 256, 256, 0, stream>>>(hbar,
      (const float*)d_in[30], (const float*)d_in[31],
      (const float*)d_in[32], (const float*)d_in[33],
      (const float*)d_in[34], (const float*)d_in[35], bfr, aex);
  k_excite<<<(NEXC * Bn + 255) / 256, 256, 0, stream>>>(aex, (const float*)d_in[36], excBuf);
  k_ks<<<Bn, 128, 0, stream>>>(bfr, excBuf, (float*)d_out);
}
